// GraphSAGELayer_5059471475381
// MI455X (gfx1250) — matrix operand order repaired and checked by reading, not compiled, not run on hardware
//
#include <hip/hip_runtime.h>
#include <hip/hip_bf16.h>

// GraphSAGE: B=32, N=1024, F_IN=F_OUT=256
// prep:    xT = transpose(x) as bf16 (per batch, [F][N]); WT = W^T bf16 [256][512]
// stage 1: h_neigh = (adj @ x) * 1/(rowsum(adj)+eps)  — adj streamed from HBM once,
//          x-chunks staged to LDS via GLOBAL_LOAD_ASYNC_TO_LDS_B128 (double-buffered)
// stage 2: out = relu([x | h_neigh] @ W + bias)
// Both GEMMs: v_wmma_f32_16x16x32_bf16, f32 accumulate.

#define NB   32
#define NN   1024
#define FIN  256
#define FOUT 256
#define EPSV 1e-8f
#define PAD  40          // LDS row stride in bf16 elems (80B, 16B-aligned, conflict-free)

typedef __attribute__((ext_vector_type(16))) __bf16  bf16x16;
typedef __attribute__((ext_vector_type(8)))  float   f32x8;

static __device__ __forceinline__ unsigned short bf_bits(float f) {
    __bf16 h = (__bf16)f;                        // native cvt, RNE
    return __builtin_bit_cast(unsigned short, h);
}

static __device__ __forceinline__ void packf(bf16x16& a, int base, float4 p, float4 q) {
    a[base + 0] = (__bf16)p.x; a[base + 1] = (__bf16)p.y;
    a[base + 2] = (__bf16)p.z; a[base + 3] = (__bf16)p.w;
    a[base + 4] = (__bf16)q.x; a[base + 5] = (__bf16)q.y;
    a[base + 6] = (__bf16)q.z; a[base + 7] = (__bf16)q.w;
}

// 16 bf16 (32B, 16B-aligned) -> fragment, pure bit reinterpretation
static __device__ __forceinline__ bf16x16 ld_bf16x16(const unsigned short* p) {
    struct U { uint4 a, b; } s;
    s.a = *(const uint4*)p;
    s.b = *(const uint4*)(p + 8);
    return __builtin_bit_cast(bf16x16, s);
}

// -------- prep A: x [B,N,F] fp32 -> xT [B,F,N] bf16 (tiled LDS transpose) --------
__global__ __launch_bounds__(256)
void sage_prep_xt(const float* __restrict__ x, unsigned short* __restrict__ xT) {
    __shared__ unsigned short t[64][65];
    const int b  = blockIdx.z;
    const int n0 = blockIdx.y * 64;
    const int f0 = blockIdx.x * 64;
    const int tf = threadIdx.x & 63;
    const int tr = threadIdx.x >> 6;              // 0..3

    const float* xb = x + ((size_t)b * NN + n0) * FIN + f0;
#pragma unroll
    for (int i = 0; i < 16; ++i) {
        int n = tr + i * 4;
        t[tf][n] = bf_bits(xb[(size_t)n * FIN + tf]);   // t[f_local][n_local]
    }
    __syncthreads();
    unsigned short* xtb = xT + ((size_t)b * FIN + f0) * NN + n0;
#pragma unroll
    for (int i = 0; i < 16; ++i) {
        int fr = tr + i * 4;
        xtb[(size_t)fr * NN + tf] = t[fr][tf];
    }
}

// -------- prep B: W[512,256] -> WT bf16 [256][512] --------
__global__ void sage_prep_wt(const float* __restrict__ weight,
                             unsigned short* __restrict__ WT) {
    int idx = blockIdx.x * blockDim.x + threadIdx.x;   // 131072
    int o  = idx >> 9;
    int kk = idx & 511;
    WT[idx] = bf_bits(weight[kk * FOUT + o]);
}

// async copy of one 32xK-chunk (16KB) of xT into LDS buffer (2 B128 per thread)
static __device__ __forceinline__ void async_copy_chunk(
        const unsigned short* __restrict__ xTb, unsigned ldsbase, int tid, int kchunk) {
#pragma unroll
    for (int j = 0; j < 2; ++j) {
        int u = tid + 512 * j;                 // 0..1023 transfer units
        int f = u >> 2;                        // 4 x 16B per f-row (32 bf16)
        int part = u & 3;
        const unsigned short* g = xTb + ((size_t)f << 10) + kchunk + part * 8;
        unsigned long long ga = (unsigned long long)g;
        unsigned loff = ldsbase + (unsigned)((f * PAD + part * 8) * 2);
        asm volatile("global_load_async_to_lds_b128 %0, %1, off"
                     :: "v"(loff), "v"(ga)
                     : "memory");
    }
}

// -------- stage 1: fused normalize + aggregate --------
// grid (8, 32): x = 128-row block, y = batch. 512 threads = 16 waves.
// wave w: Mtile = w>>1, N-half = w&1 (8 tiles of 16 cols each).
__global__ __launch_bounds__(512)
void sage_aggregate(const float* __restrict__ adj,
                    const unsigned short* __restrict__ xT,
                    unsigned short* __restrict__ hN) {
    __shared__ unsigned short ldsX[2][256 * PAD];
    __shared__ float rowsumLDS[128];

    const int bat      = blockIdx.y;
    const int rowBlock = blockIdx.x * 128;
    const int tid    = threadIdx.x;
    const int wave   = tid >> 5;
    const int lane   = tid & 31;
    const int laneLo = lane & 15;
    const int laneHi = lane >> 4;
    const int mtl    = wave >> 1;
    const int nbase  = (wave & 1) * 8;

    if (tid < 128) rowsumLDS[tid] = 0.0f;

    const unsigned short* xTb = xT + (size_t)bat * FIN * NN;
    const float* adjRow = adj + (size_t)bat * NN * NN
                              + (size_t)(rowBlock + mtl * 16 + laneLo) * NN;
    const unsigned ldsbase = (unsigned)(unsigned long long)&ldsX[0][0];
    const unsigned bufBytes = 256 * PAD * 2;

    f32x8 acc[8];
#pragma unroll
    for (int t = 0; t < 8; ++t) acc[t] = (f32x8){0,0,0,0,0,0,0,0};
    float rsum = 0.0f;

    async_copy_chunk(xTb, ldsbase, tid, 0);          // prefetch chunk 0

    for (int k = 0; k < NN; k += 32) {
        const int buf = (k >> 5) & 1;
        asm volatile("s_wait_asynccnt 0x0" ::: "memory");   // my cur-buffer loads done
        __syncthreads();                                    // all loads + prev consume done
        if (k + 32 < NN)
            async_copy_chunk(xTb, ldsbase + (buf ^ 1) * bufBytes, tid, k + 32);

        // A fragment: 16x32 bf16 from adj (fp32), fp32 row-sum on the side
        int kb = k + laneHi * 16;
        float4 a0 = *(const float4*)(adjRow + kb);
        float4 a1 = *(const float4*)(adjRow + kb + 4);
        float4 a2 = *(const float4*)(adjRow + kb + 8);
        float4 a3 = *(const float4*)(adjRow + kb + 12);
        rsum += a0.x + a0.y + a0.z + a0.w + a1.x + a1.y + a1.z + a1.w
              + a2.x + a2.y + a2.z + a2.w + a3.x + a3.y + a3.z + a3.w;
        bf16x16 afrag;
        packf(afrag, 0, a0, a1);
        packf(afrag, 8, a2, a3);

        // 8 B fragments from LDS (2x ds_load_b128 each), 8 WMMAs (A reused)
        const int kbB = laneHi * 16;
        const unsigned short* lbuf = &ldsX[buf][0];
#pragma unroll
        for (int t = 0; t < 8; ++t) {
            int col = (nbase + t) * 16 + laneLo;
            bf16x16 bfrag = ld_bf16x16(lbuf + col * PAD + kbB);
            acc[t] = __builtin_amdgcn_wmma_f32_16x16x32_bf16(
                false, afrag, false, bfrag, (short)0, acc[t], false, false);
        }
        // no trailing barrier: next overwrite targets other buffer, guarded by next top barrier
    }

    // fp32 row-sums (lanes L and L+16 hold complementary K subsets)
    if ((wave & 1) == 0)
        atomicAdd(&rowsumLDS[mtl * 16 + laneLo], rsum);
    __syncthreads();

    float inv[8];
#pragma unroll
    for (int r = 0; r < 8; ++r)
        inv[r] = 1.0f / (rowsumLDS[mtl * 16 + laneHi * 8 + r] + EPSV);

    unsigned short* hNB = hN + (size_t)bat * NN * FIN;
#pragma unroll
    for (int t = 0; t < 8; ++t) {
        int col = (nbase + t) * 16 + laneLo;
#pragma unroll
        for (int r = 0; r < 8; ++r) {
            int rowInBatch = rowBlock + mtl * 16 + laneHi * 8 + r;
            hNB[(size_t)rowInBatch * FIN + col] = bf_bits(acc[t][r] * inv[r]);
        }
    }
}

// -------- stage 2: out = relu([x | hN] @ W + bias) --------
// grid 256, block 512 (16 waves). Rows flat [B*N, F]. K = 512.
__global__ __launch_bounds__(512)
void sage_linear(const float* __restrict__ x,
                 const unsigned short* __restrict__ hN,
                 const unsigned short* __restrict__ WT,
                 const float* __restrict__ bias,
                 float* __restrict__ out) {
    const int tid    = threadIdx.x;
    const int wave   = tid >> 5;
    const int lane   = tid & 31;
    const int laneLo = lane & 15;
    const int laneHi = lane >> 4;
    const int mtl    = wave >> 1;
    const int nbase  = (wave & 1) * 8;
    const int rowbase = blockIdx.x * 128 + mtl * 16;

    f32x8 acc[8];
#pragma unroll
    for (int t = 0; t < 8; ++t) acc[t] = (f32x8){0,0,0,0,0,0,0,0};

    const int rowA = rowbase + laneLo;
    const float*          xr = x  + (size_t)rowA * FIN;
    const unsigned short* hr = hN + (size_t)rowA * FIN;

    for (int k = 0; k < 512; k += 32) {
        bf16x16 afrag;
        if (k < 256) {                         // first half of concat: x (fp32 -> bf16)
            int kb = k + laneHi * 8;
            float4 a0 = *(const float4*)(xr + kb);
            float4 a1 = *(const float4*)(xr + kb + 4);
            float4 a2 = *(const float4*)(xr + kb + 16);
            float4 a3 = *(const float4*)(xr + kb + 20);
            packf(afrag, 0, a0, a1);
            packf(afrag, 8, a2, a3);
        } else {                               // second half: h_neigh (bf16)
            int kb = (k - 256) + laneHi * 8;
            afrag = ld_bf16x16(hr + kb);       // elems 0..7 @kb, 8..15 @kb+8
            bf16x16 hi = ld_bf16x16(hr + kb + 8);  // overlapping trick avoided: rebuild
            // correct split: low 8 = [kb..kb+7], high 8 = [kb+16..kb+23]
            struct U { uint4 a, b; } s;
            s.a = *(const uint4*)(hr + kb);
            s.b = *(const uint4*)(hr + kb + 16);
            afrag = __builtin_bit_cast(bf16x16, s);
            (void)hi;
        }

        const int kbB = k + laneHi * 8;
#pragma unroll
        for (int t = 0; t < 8; ++t) {
            int o = (nbase + t) * 16 + laneLo;
            struct UB { uint4 a, b; } sb;
            sb.a = *(const uint4*)(WT + (size_t)o * 512 + kbB);
            sb.b = *(const uint4*)(WT + (size_t)o * 512 + kbB + 16);
            bf16x16 bfrag = __builtin_bit_cast(bf16x16, sb);
            acc[t] = __builtin_amdgcn_wmma_f32_16x16x32_bf16(
                false, afrag, false, bfrag, (short)0, acc[t], false, false);
        }
    }

#pragma unroll
    for (int t = 0; t < 8; ++t) {
        int col = (nbase + t) * 16 + laneLo;
        float bv = bias[col];
#pragma unroll
        for (int r = 0; r < 8; ++r) {
            int row = rowbase + laneHi * 8 + r;
            float v = acc[t][r] + bv;
            out[(size_t)row * FOUT + col] = v > 0.0f ? v : 0.0f;
        }
    }
}

extern "C" void kernel_launch(void* const* d_in, const int* in_sizes, int n_in,
                              void* d_out, int out_size, void* d_ws, size_t ws_size,
                              hipStream_t stream) {
    const float* x      = (const float*)d_in[0];   // [32,1024,256]
    const float* adj    = (const float*)d_in[1];   // [32,1024,1024]
    const float* weight = (const float*)d_in[2];   // [512,256]
    const float* bias   = (const float*)d_in[3];   // [256]
    float* out = (float*)d_out;                    // [32,1024,256]

    // ws: xT bf16 [32*256*1024] | WT bf16 [256*512] | hN bf16 [32*1024*256]  (~34 MB)
    unsigned short* xT = (unsigned short*)d_ws;
    unsigned short* WT = (unsigned short*)((char*)d_ws + (size_t)NB * FIN * NN * 2);
    unsigned short* hN = (unsigned short*)((char*)d_ws + (size_t)NB * FIN * NN * 2
                                                        + (size_t)FIN * 512 * 2);

    dim3 gX(4, 16, 32);                            // F/64, N/64, B
    sage_prep_xt<<<gX, 256, 0, stream>>>(x, xT);
    sage_prep_wt<<<512, 256, 0, stream>>>(weight, WT);

    dim3 g1(8, 32);                                // 128-row blocks x batches
    sage_aggregate<<<g1, 512, 0, stream>>>(adj, xT, hN);

    sage_linear<<<256, 512, 0, stream>>>(x, hN, WT, bias, out);
}
